// GatGnn_26027501814367
// MI455X (gfx1250) — compile-verified, matrix operand order checked by reading, not run on hardware
//
#include <hip/hip_runtime.h>
#include <hip/hip_bf16.h>

// ---------------------------------------------------------------------------
// GAT-GNN pipeline for MI455X (gfx1250, wave32, WMMA).
// Dense GEMMs (K=128) use v_wmma_f32_16x16x32_f16: one wave computes a 32x64
// output block (8 accumulators) so each B-tile load feeds 2 WMMAs and each
// A-tile load feeds 4 (1.5 b128 loads per WMMA).
// Edge softmax/scatter is the HBM-bound part (~435MB gather + 109M atomics per
// step); feature loops are lane-strided (stride 32) for coalescing.
// ---------------------------------------------------------------------------

typedef __attribute__((ext_vector_type(16))) _Float16 v16h;
typedef __attribute__((ext_vector_type(8)))  _Float16 v8h;
typedef __attribute__((ext_vector_type(8)))  float    v8f;

union U16h { v16h v; v8h h[2]; };

__device__ __forceinline__ float leakyf(float v, float s) { return v >= 0.f ? v : s * v; }

__device__ __forceinline__ float fast_tanh(float x) {
  // tanh(x) = 1 - 2/(exp(2x)+1)
  return 1.f - 2.f / (__expf(2.f * x) + 1.f);
}

__device__ __forceinline__ void atomicMaxF(float* addr, float val) {
  unsigned int* ua = (unsigned int*)addr;
  unsigned int cur = *ua;
  while (__uint_as_float(cur) < val) {
    unsigned int prev = atomicCAS(ua, cur, __float_as_uint(val));
    if (prev == cur) break;
    cur = prev;
  }
}

// -------------------- weight preparation --------------------
__global__ void k_cvt_f16(const float* __restrict__ in, _Float16* __restrict__ out, int n) {
  int i = blockIdx.x * blockDim.x + threadIdx.x;
  if (i < n) out[i] = (_Float16)in[i];
}

// out[c*rows + r] = in[r*cols + c]  (f32 -> f16 transpose; W2 (HxH) -> W2^T)
__global__ void k_transpose_cvt(const float* __restrict__ in, _Float16* __restrict__ out,
                                int rows, int cols) {
  int i = blockIdx.x * blockDim.x + threadIdx.x;
  if (i >= rows * cols) return;
  int c = i / rows, r = i % rows;
  out[i] = (_Float16)in[r * cols + c];
}

__global__ void k_zero1(float* p) { *p = 0.f; }

// -------------------- WMMA GEMM: C[MxNout] = A[MxK] * Bt[Nout x K] ----------
// A, Bt row-major f16 (both K-contiguous). K % 32 == 0, Nout % 64 == 0,
// M % 16 == 0; A buffer must be padded to a multiple of 32 rows (pad rows are
// read but their outputs are store-guarded). One wave -> 32x64 block.
__global__ __launch_bounds__(128) void k_wmma_gemm(
    const _Float16* __restrict__ A, const _Float16* __restrict__ Bt,
    float* __restrict__ C, int M, int K, int Nout) {
  const int wave = threadIdx.x >> 5;
  const int lane = threadIdx.x & 31;
  const int nT4  = Nout >> 6;                 // strips of 4 N-tiles
  const int m32  = (M + 31) >> 5;             // 32-row blocks (last may be half)
  const int tiles = m32 * nT4;
  const int tid = blockIdx.x * 4 + wave;      // wave-uniform -> EXEC stays full
  if (tid >= tiles) return;
  const int tm  = tid / nT4;
  const int tn4 = tid % nT4;
  const int row = lane & 15;
  const int kh  = (lane >> 4) << 3;           // 0 or 8 (wave32 A/B layout)

  const _Float16* arow0 = A  + (size_t)(tm * 32 + row) * K;
  const _Float16* arow1 = arow0 + (size_t)16 * K;
  const _Float16* bbas  = Bt + (size_t)(tn4 * 64 + row) * K;

  v8f acc[2][4] = {{v8f{}, v8f{}, v8f{}, v8f{}}, {v8f{}, v8f{}, v8f{}, v8f{}}};
  for (int k = 0; k < K; k += 32) {
    U16h a0, a1;
    a0.h[0] = *(const v8h*)(arow0 + k + kh);
    a0.h[1] = *(const v8h*)(arow0 + k + 16 + kh);
    a1.h[0] = *(const v8h*)(arow1 + k + kh);
    a1.h[1] = *(const v8h*)(arow1 + k + 16 + kh);
#pragma unroll
    for (int nt = 0; nt < 4; ++nt) {
      const _Float16* brow = bbas + (size_t)(nt * 16) * K;
      U16h b;
      b.h[0] = *(const v8h*)(brow + k + kh);
      b.h[1] = *(const v8h*)(brow + k + 16 + kh);
      acc[0][nt] = __builtin_amdgcn_wmma_f32_16x16x32_f16(
          false, a0.v, false, b.v, (short)0, acc[0][nt], false, false);
      acc[1][nt] = __builtin_amdgcn_wmma_f32_16x16x32_f16(
          false, a1.v, false, b.v, (short)0, acc[1][nt], false, false);
    }
  }
  const int roff = (lane >> 4) * 8;           // C/D layout: vgpr r -> row r (+8 hi half)
#pragma unroll
  for (int mt = 0; mt < 2; ++mt) {
    const int rbase = tm * 32 + mt * 16;
    if (rbase >= M) continue;                 // pad block (M % 16 == 0 => whole tile valid)
#pragma unroll
    for (int nt = 0; nt < 4; ++nt) {
      const int col = tn4 * 64 + nt * 16 + (lane & 15);
#pragma unroll
      for (int r = 0; r < 8; ++r)
        C[(size_t)(rbase + roff + r) * Nout + col] = acc[mt][nt][r];
    }
  }
}

// -------------------- small-K (K=2) GEMMs --------------------
// out[i][o] = x[2i]*W[o*so] + x[2i+1]*W[o*so+sk]
__global__ void k_k2gemm(const float* __restrict__ x, const float* __restrict__ W,
                         float* __restrict__ out, int Nn, int Nout, int so, int sk) {
  long long idx = (long long)blockIdx.x * blockDim.x + threadIdx.x;
  if (idx >= (long long)Nn * Nout) return;
  int i = (int)(idx / Nout), o = (int)(idx % Nout);
  out[idx] = x[2 * i] * W[o * so] + x[2 * i + 1] * W[o * so + sk];
}

// -------------------- attention scores: one wave per node --------------------
__global__ void k_scores(const float* __restrict__ Hm, const float* __restrict__ asrc,
                         const float* __restrict__ adst, float* __restrict__ ssrc,
                         float* __restrict__ sdst, int Nn, int H) {
  int w = (int)(((long long)blockIdx.x * blockDim.x + threadIdx.x) >> 5);
  int lane = threadIdx.x & 31;
  if (w >= Nn) return;
  const float* rowp = Hm + (size_t)w * H;
  float p1 = 0.f, p2 = 0.f;
  for (int f = lane; f < H; f += 32) { float v = rowp[f]; p1 += v * asrc[f]; p2 += v * adst[f]; }
#pragma unroll
  for (int off = 16; off > 0; off >>= 1) {
    p1 += __shfl_down(p1, off, 32);
    p2 += __shfl_down(p2, off, 32);
  }
  if (lane == 0) { ssrc[w] = p1; sdst[w] = p2; }
}

// -------------------- edge softmax + scatter --------------------
__global__ void k_gat_init(float* __restrict__ out, float* __restrict__ m,
                           float* __restrict__ den, const float* __restrict__ bias,
                           int Nn, int H) {
  long long idx = (long long)blockIdx.x * blockDim.x + threadIdx.x;
  if (idx >= (long long)Nn * H) return;
  out[idx] = bias[idx % H];
  if (idx < Nn) { m[idx] = -3.0e38f; den[idx] = 0.f; }
}

__device__ __forceinline__ void edge_sd(const int* ei, int e, int E, int& s, int& d) {
  if (e < E) { s = ei[e]; d = ei[E + e]; } else { s = e - E; d = s; }
}

__global__ void k_edge_logit(const int* __restrict__ ei, const float* __restrict__ ssrc,
                             const float* __restrict__ sdst, float* __restrict__ ev,
                             float* __restrict__ m, int E, int Nn) {
  int e = blockIdx.x * blockDim.x + threadIdx.x;
  if (e >= E + Nn) return;
  int s, d; edge_sd(ei, e, E, s, d);
  float v = leakyf(ssrc[s] + sdst[d], 0.2f);
  ev[e] = v;
  atomicMaxF(m + d, v);
}

__global__ void k_edge_exp(const int* __restrict__ ei, const float* __restrict__ ev,
                           const float* __restrict__ m, float* __restrict__ ex,
                           float* __restrict__ den, int E, int Nn) {
  int e = blockIdx.x * blockDim.x + threadIdx.x;
  if (e >= E + Nn) return;
  int s, d; edge_sd(ei, e, E, s, d);
  float v = __expf(ev[e] - m[d]);
  ex[e] = v;
  atomicAdd(den + d, v);
}

// one wave per edge, lanes stride features (coalesced atomics)
__global__ void k_edge_scatter(const int* __restrict__ ei, const float* __restrict__ ex,
                               const float* __restrict__ den, const float* __restrict__ Hm,
                               float* __restrict__ out, int E, int Nn, int H) {
  long long t = (long long)blockIdx.x * blockDim.x + threadIdx.x;
  int e = (int)(t >> 5);
  if (e >= E + Nn) return;
  int lane = (int)(t & 31);
  int s, d; edge_sd(ei, e, E, s, d);
  float alpha = ex[e] / den[d];
  const float* hs = Hm + (size_t)s * H;
  float* od = out + (size_t)d * H;
  for (int f = lane; f < H; f += 32) atomicAdd(od + f, hs[f] * alpha);
}

// -------------------- GRU combine (gate order r,z,n) --------------------
__global__ void k_gru_combine(const float* __restrict__ gi, const float* __restrict__ gh,
                              const float* __restrict__ hprev,
                              const float* __restrict__ bih, const float* __restrict__ bhh,
                              float* __restrict__ state, _Float16* __restrict__ state_h,
                              int Nn, int H) {
  long long idx = (long long)blockIdx.x * blockDim.x + threadIdx.x;
  if (idx >= (long long)Nn * H) return;
  int i = (int)(idx / H), j = (int)(idx % H);
  size_t b = (size_t)i * 3 * H;
  float gir = gi[b + j] + bih[j],           ghr = gh[b + j] + bhh[j];
  float giz = gi[b + H + j] + bih[H + j],   ghz = gh[b + H + j] + bhh[H + j];
  float gin = gi[b + 2 * H + j] + bih[2 * H + j], ghn = gh[b + 2 * H + j] + bhh[2 * H + j];
  float r = 1.f / (1.f + __expf(-(gir + ghr)));
  float z = 1.f / (1.f + __expf(-(giz + ghz)));
  float nn = fast_tanh(gin + r * ghn);
  float o = (1.f - z) * nn + z * hprev[idx];
  state[idx] = o;
  state_h[idx] = (_Float16)o;
}

// -------------------- f32 -> f16 node features --------------------
__global__ void k_cvt_rows(const float* __restrict__ in, _Float16* __restrict__ out, long long n) {
  long long i = (long long)blockIdx.x * blockDim.x + threadIdx.x;
  if (i < n) out[i] = (_Float16)in[i];
}

// -------------------- readout MLP (H->16->8->3, leaky 0.01) + MSE -----------
__global__ void k_mlp_loss(const float* __restrict__ state,
                           const float* __restrict__ W1, const float* __restrict__ b1,
                           const float* __restrict__ W2, const float* __restrict__ b2,
                           const float* __restrict__ W3, const float* __restrict__ b3,
                           const float* __restrict__ target, float* __restrict__ outp,
                           float* __restrict__ lossAcc, int Nn, int H) {
  int i = blockIdx.x * blockDim.x + threadIdx.x;
  if (i >= Nn) return;
  const float* row = state + (size_t)i * H;
  float h1[16];
#pragma unroll
  for (int u = 0; u < 16; ++u) h1[u] = b1[u];
  for (int k = 0; k < H; ++k) {
    float sv = row[k];
    const float* wr = W1 + k * 16;
#pragma unroll
    for (int u = 0; u < 16; ++u) h1[u] += sv * wr[u];
  }
#pragma unroll
  for (int u = 0; u < 16; ++u) h1[u] = leakyf(h1[u], 0.01f);
  float h2[8];
#pragma unroll
  for (int u = 0; u < 8; ++u) h2[u] = b2[u];
#pragma unroll
  for (int k = 0; k < 16; ++k)
#pragma unroll
    for (int u = 0; u < 8; ++u) h2[u] += h1[k] * W2[k * 8 + u];
#pragma unroll
  for (int u = 0; u < 8; ++u) h2[u] = leakyf(h2[u], 0.01f);
  float o[3];
#pragma unroll
  for (int u = 0; u < 3; ++u) o[u] = b3[u];
#pragma unroll
  for (int k = 0; k < 8; ++k)
#pragma unroll
    for (int u = 0; u < 3; ++u) o[u] += h2[k] * W3[k * 3 + u];
  float le = 0.f;
#pragma unroll
  for (int u = 0; u < 3; ++u) {
    float v = leakyf(o[u], 0.01f);
    outp[(size_t)i * 3 + u] = v;
    float dd = v - target[(size_t)i * 3 + u];
    le += dd * dd;
  }
  atomicAdd(lossAcc, le);
}

__global__ void k_finalize_loss(const float* __restrict__ acc, float* __restrict__ dst, float inv) {
  *dst = (*acc) * inv;
}

// ---------------------------------------------------------------------------
extern "C" void kernel_launch(void* const* d_in, const int* in_sizes, int n_in,
                              void* d_out, int out_size, void* d_ws, size_t ws_size,
                              hipStream_t stream) {
  const float* x       = (const float*)d_in[0];
  const int*   ei      = (const int*)d_in[1];
  const float* targetn = (const float*)d_in[2];
  const float* W1      = (const float*)d_in[4];
  const float* a_src1  = (const float*)d_in[5];
  const float* a_dst1  = (const float*)d_in[6];
  const float* b1      = (const float*)d_in[7];
  const float* W2      = (const float*)d_in[8];
  const float* a_src2  = (const float*)d_in[9];
  const float* a_dst2  = (const float*)d_in[10];
  const float* b2      = (const float*)d_in[11];
  const float* g1Wih   = (const float*)d_in[12];
  const float* g1Whh   = (const float*)d_in[13];
  const float* g1bih   = (const float*)d_in[14];
  const float* g1bhh   = (const float*)d_in[15];
  const float* gWih    = (const float*)d_in[16];
  const float* gWhh    = (const float*)d_in[17];
  const float* gbih    = (const float*)d_in[18];
  const float* gbhh    = (const float*)d_in[19];
  const float* mW1     = (const float*)d_in[20];
  const float* mb1     = (const float*)d_in[21];
  const float* mW2     = (const float*)d_in[22];
  const float* mb2     = (const float*)d_in[23];
  const float* mW3     = (const float*)d_in[24];
  const float* mb3     = (const float*)d_in[25];

  const int Nn = in_sizes[0] / 2;   // 50000
  const int E  = in_sizes[1] / 2;   // 800000
  const int H  = in_sizes[5];       // 128
  const int H3 = 3 * H;
  const int Et = E + Nn;
  const int Mpad = (Nn + 31) & ~31; // pad A-side f16 buffers to 32-row multiple

  // bump allocator over d_ws (256B aligned regions)
  char* base = (char*)d_ws;
  size_t off = 0;
  auto alloc = [&](size_t bytes) -> void* {
    void* p = base + off;
    off = (off + bytes + 255) & ~(size_t)255;
    return p;
  };
  float*     state   = (float*)alloc((size_t)Nn * H * 4);
  _Float16*  state_h = (_Float16*)alloc((size_t)Mpad * H * 2);
  float*     Hm      = (float*)alloc((size_t)Nn * H * 4);   // GAT hidden h = state@W
  float*     Y       = (float*)alloc((size_t)Nn * H * 4);   // GAT output
  _Float16*  Yh      = (_Float16*)alloc((size_t)Mpad * H * 2);
  float*     gi      = (float*)alloc((size_t)Nn * H3 * 4);
  float*     gh      = (float*)alloc((size_t)Nn * H3 * 4);
  float*     ssrc    = (float*)alloc((size_t)Nn * 4);
  float*     sdst    = (float*)alloc((size_t)Nn * 4);
  float*     mseg    = (float*)alloc((size_t)Nn * 4);
  float*     dseg    = (float*)alloc((size_t)Nn * 4);
  float*     ev      = (float*)alloc((size_t)Et * 4);
  float*     exw     = (float*)alloc((size_t)Et * 4);
  _Float16*  W2t     = (_Float16*)alloc((size_t)H * H * 2);
  _Float16*  g1Whh_h = (_Float16*)alloc((size_t)H3 * H * 2);
  _Float16*  gWih_h  = (_Float16*)alloc((size_t)H3 * H * 2);
  _Float16*  gWhh_h  = (_Float16*)alloc((size_t)H3 * H * 2);
  float*     lossAcc = (float*)alloc(4);

  auto blk = [](long long n, int b) { return (unsigned)((n + b - 1) / b); };

  // ---- weight prep (every call; deterministic, no caching) ----
  k_transpose_cvt<<<blk((long long)H * H, 256), 256, 0, stream>>>(W2, W2t, H, H);
  k_cvt_f16<<<blk((long long)H3 * H, 256), 256, 0, stream>>>(g1Whh, g1Whh_h, H3 * H);
  k_cvt_f16<<<blk((long long)H3 * H, 256), 256, 0, stream>>>(gWih, gWih_h, H3 * H);
  k_cvt_f16<<<blk((long long)H3 * H, 256), 256, 0, stream>>>(gWhh, gWhh_h, H3 * H);
  k_zero1<<<1, 1, 0, stream>>>(lossAcc);

  auto gat_edges = [&](const float* src_feat, const float* asrc, const float* adst,
                       const float* bias) {
    k_scores<<<blk((long long)Nn * 32, 256), 256, 0, stream>>>(src_feat, asrc, adst, ssrc, sdst, Nn, H);
    k_gat_init<<<blk((long long)Nn * H, 256), 256, 0, stream>>>(Y, mseg, dseg, bias, Nn, H);
    k_edge_logit<<<blk(Et, 256), 256, 0, stream>>>(ei, ssrc, sdst, ev, mseg, E, Nn);
    k_edge_exp<<<blk(Et, 256), 256, 0, stream>>>(ei, ev, mseg, exw, dseg, E, Nn);
    k_edge_scatter<<<blk((long long)Et * 32, 256), 256, 0, stream>>>(ei, exw, dseg, src_feat, Y, E, Nn, H);
  };
  auto gemm = [&](const _Float16* A, const _Float16* Bt, float* C, int M, int K, int Nout) {
    int tiles = ((M + 31) / 32) * (Nout / 64);
    k_wmma_gemm<<<blk(tiles, 4), 128, 0, stream>>>(A, Bt, C, M, K, Nout);
  };

  // ---- layer 1: GAT(x; W1) then GRU1(x, gat_out) ----
  k_k2gemm<<<blk((long long)Nn * H, 256), 256, 0, stream>>>(x, W1, Hm, Nn, H, 1, H);
  gat_edges(Hm, a_src1, a_dst1, b1);
  k_k2gemm<<<blk((long long)Nn * H3, 256), 256, 0, stream>>>(x, g1Wih, gi, Nn, H3, 2, 1);
  k_cvt_rows<<<blk((long long)Nn * H, 256), 256, 0, stream>>>(Y, Yh, (long long)Nn * H);
  gemm(Yh, g1Whh_h, gh, Nn, H, H3);
  k_gru_combine<<<blk((long long)Nn * H, 256), 256, 0, stream>>>(gi, gh, Y, g1bih, g1bhh,
                                                                 state, state_h, Nn, H);

  // ---- propagation steps 2..4: GAT(state; W2) then GRU(state, gat_out) ----
  for (int t = 0; t < 3; ++t) {
    gemm(state_h, W2t, Hm, Nn, H, H);
    gat_edges(Hm, a_src2, a_dst2, b2);
    k_cvt_rows<<<blk((long long)Nn * H, 256), 256, 0, stream>>>(Y, Yh, (long long)Nn * H);
    gemm(state_h, gWih_h, gi, Nn, H, H3);
    gemm(Yh, gWhh_h, gh, Nn, H, H3);
    k_gru_combine<<<blk((long long)Nn * H, 256), 256, 0, stream>>>(gi, gh, Y, gbih, gbhh,
                                                                   state, state_h, Nn, H);
  }

  // ---- readout + loss ----
  float* outp = (float*)d_out;
  k_mlp_loss<<<blk(Nn, 128), 128, 0, stream>>>(state, mW1, mb1, mW2, mb2, mW3, mb3,
                                               targetn, outp, lossAcc, Nn, H);
  k_finalize_loss<<<1, 1, 0, stream>>>(lossAcc, outp + (size_t)Nn * 3,
                                       1.f / (float)(Nn * 3));
}
